// DGCNNNet_2860448219406
// MI455X (gfx1250) — matrix operand order repaired and checked by reading, not compiled, not run on hardware
//
#include <hip/hip_runtime.h>
#include <math.h>

#define BB 32
#define PP 2048
#define KNN 20
#define MEDGE ((size_t)BB * PP * KNN)   // 1,310,720 edge rows
#define BP ((size_t)BB * PP)            // 65,536 point rows
#define EPSB 1e-5f

typedef __attribute__((ext_vector_type(16))) __bf16 v16bf;
typedef __attribute__((ext_vector_type(8)))  float  v8f;

// ---------------------------------------------------------------------------
// WMMA fragment loaders (CDNA5 16x16x32 bf16 layouts, ISA 7.12.2)
// A 16x32: lanes0-15 rows M, VGPR0-3: K=0..7 (lanes0-15) / K=8..15 (lanes16-31),
//          VGPR4-7: K=16..23 / 24..31, two bf16 per VGPR.
// B 32x16: VGPR0-3: K=0..7 (lanes0-15) / K=8..15 (lanes16-31), VGPR4-7: K=16..23 / 24..31 ; N=lane%16.
// C/D 16x16 f32: VGPR r: M=r (lanes0-15) / M=8+r (lanes16-31), N=lane%16.
// ---------------------------------------------------------------------------
__device__ __forceinline__ v16bf load_a_frag_lds(const __bf16* as, int lda,
                                                 int lane, int mbase, int kbase) {
  v16bf a;
  int m = mbase + (lane & 15);
  int khalf = (lane < 16) ? 0 : 8;
#pragma unroll
  for (int j = 0; j < 8; ++j) {
    int k0 = kbase + ((j < 4) ? 0 : 16) + khalf + 2 * (j & 3);
    a[2 * j]     = as[m * lda + k0];
    a[2 * j + 1] = as[m * lda + k0 + 1];
  }
  return a;
}

__device__ __forceinline__ v16bf load_a_frag_f32(const float* X, int lda,
                                                 int lane, int mbase, int kbase) {
  v16bf a;
  int m = mbase + (lane & 15);
  int khalf = (lane < 16) ? 0 : 8;
#pragma unroll
  for (int j = 0; j < 8; ++j) {
    int k0 = kbase + ((j < 4) ? 0 : 16) + khalf + 2 * (j & 3);
    a[2 * j]     = (__bf16)X[(size_t)m * lda + k0];
    a[2 * j + 1] = (__bf16)X[(size_t)m * lda + k0 + 1];
  }
  return a;
}

// pre-swizzled weight fragments: Wfrag[(kt*NT + nt)*512 + lane*16 + t]
__device__ __forceinline__ v16bf load_b_fragS(const __bf16* __restrict__ Wfrag,
                                              int ntn, int lane, int kt, int nt) {
  const v16bf* p =
      (const v16bf*)(Wfrag + (((size_t)kt * ntn + nt) * 512) + lane * 16);
  return *p;
}

__device__ __forceinline__ v16bf load_b_frag_f32(const float* W, int ldb,
                                                 int lane, int kbase, int nbase) {
  v16bf b;
  int n = nbase + (lane & 15);
  int koff = (lane < 16) ? 0 : 8;
#pragma unroll
  for (int j = 0; j < 8; ++j) {
    int k0 = kbase + ((j < 4) ? 0 : 16) + koff + 2 * (j & 3);
    b[2 * j]     = (__bf16)W[(size_t)k0 * ldb + n];
    b[2 * j + 1] = (__bf16)W[(size_t)(k0 + 1) * ldb + n];
  }
  return b;
}

// ---------------------------------------------------------------------------
// util kernels
// ---------------------------------------------------------------------------
__global__ void fill_zero(float* p, size_t n) {
  size_t i = (size_t)blockIdx.x * 256 + threadIdx.x;
  size_t stride = (size_t)gridDim.x * 256;
  for (; i < n; i += stride) p[i] = 0.f;
}

// pairwise squared distances within each batch: D[b][p][q] = |x_p - x_q|^2
template <int C>
__global__ __launch_bounds__(256) void dist_kernel(const float* __restrict__ x,
                                                   float* __restrict__ D) {
  __shared__ float xp[16][C];
  __shared__ float xq[64][C];
  int b = blockIdx.y, pt = blockIdx.x, tid = threadIdx.x;
  for (int i = tid; i < 16 * C; i += 256)
    xp[i / C][i % C] = x[((size_t)b * PP + pt * 16 + i / C) * C + i % C];
  const float* xb = x + (size_t)b * PP * C;
  float* Drow = D + ((size_t)b * PP + pt * 16) * PP;
  int pi = tid >> 4, qi = tid & 15;
  for (int qt = 0; qt < PP / 64; ++qt) {
    __syncthreads();
    for (int i = tid; i < 64 * C; i += 256)
      xq[i / C][i % C] = xb[((size_t)qt * 64 + i / C) * C + i % C];
    __syncthreads();
#pragma unroll
    for (int j = 0; j < 4; ++j) {
      int ql = j * 16 + qi;
      float d = 0.f;
#pragma unroll
      for (int c = 0; c < C; ++c) {
        float t = xp[pi][c] - xq[ql][c];
        d += t * t;
      }
      Drow[(size_t)pi * PP + qt * 64 + ql] = d;
    }
  }
}

// top-K smallest per row via K threshold re-scans (no dynamic reg indexing)
__global__ __launch_bounds__(128) void knn_select(const float* __restrict__ D,
                                                  int* __restrict__ idx) {
  size_t p = (size_t)blockIdx.x * 128 + threadIdx.x;
  if (p >= BP) return;
  const float* row = D + p * PP;
  float lastD = -1.f;
  int lastQ = -1;
  for (int t = 0; t < KNN; ++t) {
    float bd = 3.4e38f;
    int bq = 0;
    for (int q = 0; q < PP; ++q) {
      float d = row[q];
      bool beyond = (d > lastD) || (d == lastD && q > lastQ);
      if (beyond && d < bd) { bd = d; bq = q; }
    }
    idx[p * KNN + t] = bq;
    lastD = bd;
    lastQ = bq;
  }
}

// fold previous-layer BN shift into bias: b'[o] = b[o] + sum_c t_c * W[c,o]
__global__ void prep_bias(const float* __restrict__ W, const float* __restrict__ bias,
                          const float* __restrict__ gam, const float* __restrict__ bet,
                          const float* __restrict__ stats, float cnt, int use_stats,
                          int cin, int cout, float* __restrict__ bprime) {
  int o = blockIdx.x, tid = threadIdx.x;
  __shared__ float red[64];
  float acc = 0.f;
  if (use_stats) {
    for (int c = tid; c < cin; c += 64) {
      float m = stats[c] / cnt;
      float v = stats[cin + c] / cnt - m * m;
      float s = gam[c] * rsqrtf(v + EPSB);
      float t = bet[c] - m * s;
      acc += t * W[(size_t)c * cout + o];
    }
  }
  red[tid] = acc;
  __syncthreads();
  for (int st = 32; st > 0; st >>= 1) {
    if (tid < st) red[tid] += red[tid + st];
    __syncthreads();
  }
  if (tid == 0) bprime[o] = bias[o] + red[0];
}

// scale-fold W and swizzle into B-fragment-ready layout (32B per lane per tile)
__global__ void prep_wfrag(const float* __restrict__ W, const float* __restrict__ gam,
                           const float* __restrict__ stats, float cnt, int use_stats,
                           int cin, int kpad, int cout, __bf16* __restrict__ Wfrag) {
  int ntn = cout / 16;
  size_t total = (size_t)(kpad / 32) * ntn * 512;
  size_t i = (size_t)blockIdx.x * 256 + threadIdx.x;
  size_t stride = (size_t)gridDim.x * 256;
  for (; i < total; i += stride) {
    int t = (int)(i & 15);
    int lane = (int)((i >> 4) & 31);
    int tile = (int)(i >> 9);
    int nt = tile % ntn, kt = tile / ntn;
    int koff = (lane < 16) ? 0 : 8;
    int k = kt * 32 + ((t < 8) ? 0 : 16) + koff + (t & 7);
    int n = nt * 16 + (lane & 15);
    float v = 0.f;
    if (k < cin) {
      float s = 1.f;
      if (use_stats) {
        float m = stats[k] / cnt;
        float va = stats[cin + k] / cnt - m * m;
        s = gam[k] * rsqrtf(va + EPSB);
      }
      v = s * W[(size_t)k * cout + n];
    }
    Wfrag[i] = (__bf16)v;
  }
}

// y[c] = BN(in[c]) with stats = {sum[C], sumsq[C]} over cnt rows
__global__ void bn_apply(const float* __restrict__ in, float* __restrict__ outp,
                         const float* __restrict__ stats, const float* __restrict__ gam,
                         const float* __restrict__ bet, float cnt, int C, size_t n) {
  size_t i = (size_t)blockIdx.x * 256 + threadIdx.x;
  size_t stride = (size_t)gridDim.x * 256;
  for (; i < n; i += stride) {
    int c = (int)(i % C);
    float m = stats[c] / cnt;
    float v = stats[C + c] / cnt - m * m;
    float s = gam[c] * rsqrtf(v + EPSB);
    outp[i] = (in[i] - m) * s + bet[c];
  }
}

// ---------------------------------------------------------------------------
// fused GEMM + bias + relu (+stats, + store | group-atomicMax)
// SMODE: 0 = edge feats from pos (Cin=6 padded), 1 = bf16 buffer [M][KPAD] via
//        GLOBAL_LOAD_ASYNC_TO_LDS_B128, 2 = edge feats from x1 (Cin=128),
//        3 = concat(x1[64], x2[128]) (Cin=192)
// OMODE: 0 = store bf16 y to out[M][COUT]; 1 = atomicMax into out[row/gdiv][COUT]
// ---------------------------------------------------------------------------
template <int KPAD, int COUT, int SMODE, int OMODE>
__global__ __launch_bounds__(256) void mlp_gemm(
    const float* __restrict__ X0, const float* __restrict__ X1,
    const __bf16* __restrict__ Xb, const int* __restrict__ idx,
    const __bf16* __restrict__ Wfrag, const float* __restrict__ bprime,
    void* __restrict__ outv, float* __restrict__ stats, int gdiv) {
  __shared__ __attribute__((aligned(16))) __bf16 as[128 * KPAD];
  __shared__ float sstat[2 * COUT];
  int tid = threadIdx.x;
  size_t rowbase = (size_t)blockIdx.x * 128;
  for (int i = tid; i < 2 * COUT; i += 256) sstat[i] = 0.f;

  if constexpr (SMODE == 0) {
    if (tid < 128) {
      size_t r = rowbase + tid;
      int b = (int)(r / ((size_t)PP * KNN));
      int rem = (int)(r % ((size_t)PP * KNN));
      int p = rem / KNN, k = rem % KNN;
      int j = idx[((size_t)b * PP + p) * KNN + k];
      const float* xi = X0 + ((size_t)b * PP + p) * 3;
      const float* xj = X0 + ((size_t)b * PP + j) * 3;
      __bf16* arow = as + tid * KPAD;
#pragma unroll
      for (int c = 0; c < 3; ++c) {
        arow[c]     = (__bf16)xi[c];
        arow[3 + c] = (__bf16)(xj[c] - xi[c]);
      }
      for (int c = 6; c < KPAD; ++c) arow[c] = (__bf16)0.f;
    }
  } else if constexpr (SMODE == 1) {
    // pure byte copy global(bf16) -> LDS: CDNA5 async load, ASYNCcnt tracked
    constexpr int NB = 128 * KPAD * 2;      // bytes per tile (16 KB for KPAD=64)
    unsigned ldsbase = (unsigned)(size_t)(void*)as;  // low 32 bits = LDS offset
    const char* gsrc = (const char*)(Xb + rowbase * KPAD);
#pragma unroll
    for (int it = 0; it < NB / 4096; ++it) {
      int boff = (tid + it * 256) * 16;
      unsigned laddr = ldsbase + (unsigned)boff;
      const char* gaddr = gsrc + boff;
      asm volatile("global_load_async_to_lds_b128 %0, %1, off"
                   :
                   : "v"(laddr), "v"(gaddr)
                   : "memory");
    }
    asm volatile("s_wait_asynccnt 0x0" ::: "memory");
  } else if constexpr (SMODE == 2) {
    int rl = tid >> 1, half = tid & 1;
    size_t r = rowbase + rl;
    int b = (int)(r / ((size_t)PP * KNN));
    int rem = (int)(r % ((size_t)PP * KNN));
    int p = rem / KNN, k = rem % KNN;
    int j = idx[((size_t)b * PP + p) * KNN + k];
    const float* xi = X0 + ((size_t)b * PP + p) * 64;
    const float* xj = X0 + ((size_t)b * PP + j) * 64;
    __bf16* dst = as + rl * KPAD;
    if (half == 0)
      for (int c = 0; c < 64; ++c) dst[c] = (__bf16)xi[c];
    else
      for (int c = 0; c < 64; ++c) dst[64 + c] = (__bf16)(xj[c] - xi[c]);
  } else {  // SMODE == 3
    int rl = tid >> 1, half = tid & 1;
    size_t r = rowbase + rl;
    __bf16* dst = as + rl * KPAD;
    if (half == 0) {
      const float* s1 = X0 + r * 64;
      for (int c = 0; c < 64; ++c) dst[c] = (__bf16)s1[c];
      const float* s2 = X1 + r * 128;
      for (int c = 0; c < 32; ++c) dst[64 + c] = (__bf16)s2[c];
    } else {
      const float* s2 = X1 + r * 128;
      for (int c = 32; c < 128; ++c) dst[64 + c] = (__bf16)s2[c];
    }
  }
  __syncthreads();

  int wave = tid >> 5, lane = tid & 31;
  int mbl = wave * 16;
  constexpr int NKT = KPAD / 32;
  constexpr int NTN = COUT / 16;
  v16bf afrag[NKT];
#pragma unroll
  for (int kt = 0; kt < NKT; ++kt)
    afrag[kt] = load_a_frag_lds(as, KPAD, lane, mbl, kt * 32);

  int rowoff = (lane < 16) ? 0 : 8;
  for (int nt = 0; nt < NTN; ++nt) {
    v8f acc = {0.f, 0.f, 0.f, 0.f, 0.f, 0.f, 0.f, 0.f};
#pragma unroll
    for (int kt = 0; kt < NKT; ++kt) {
      v16bf bfrag = load_b_fragS(Wfrag, NTN, lane, kt, nt);
      acc = __builtin_amdgcn_wmma_f32_16x16x32_bf16(false, afrag[kt], false, bfrag,
                                                    (short)0, acc, false, false);
    }
    int col = nt * 16 + (lane & 15);
    float bv = bprime[col];
    float s = 0.f, ss = 0.f;
#pragma unroll
    for (int r = 0; r < 8; ++r) {
      float v = acc[r] + bv;
      v = v > 0.f ? v : 0.f;
      s += v;
      ss += v * v;
      size_t row = rowbase + mbl + rowoff + r;
      if constexpr (OMODE == 0) {
        ((__bf16*)outv)[row * COUT + col] = (__bf16)v;
      } else {
        // relu output >= 0: uint bit-pattern order == float order
        atomicMax((unsigned int*)&((float*)outv)[(row / gdiv) * COUT + col],
                  __float_as_uint(v));
      }
    }
    atomicAdd(&sstat[col], s);
    atomicAdd(&sstat[COUT + col], ss);
  }
  __syncthreads();
  for (int i = tid; i < COUT; i += 256) {
    atomicAdd(&stats[i], sstat[i]);
    atomicAdd(&stats[COUT + i], sstat[COUT + i]);
  }
}

// ---------------------------------------------------------------------------
// head layer: 32 rows, single block. y = BN(relu(x@W+b)), stats over 32 rows.
// ---------------------------------------------------------------------------
template <int CIN, int COUT>
__global__ __launch_bounds__(256) void head_kernel(
    const float* __restrict__ xin, const float* __restrict__ W,
    const float* __restrict__ bias, const float* __restrict__ gam,
    const float* __restrict__ bet, float* __restrict__ out) {
  __shared__ float ys[32 * COUT];
  int tid = threadIdx.x, wave = tid >> 5, lane = tid & 31;
  constexpr int NT = COUT / 16;
  for (int t = wave; t < 2 * NT; t += 8) {
    int mt = t / NT, nt = t % NT;
    v8f acc = {0.f, 0.f, 0.f, 0.f, 0.f, 0.f, 0.f, 0.f};
#pragma unroll 4
    for (int kt = 0; kt < CIN / 32; ++kt) {
      v16bf a = load_a_frag_f32(xin, CIN, lane, mt * 16, kt * 32);
      v16bf b = load_b_frag_f32(W, COUT, lane, kt * 32, nt * 16);
      acc = __builtin_amdgcn_wmma_f32_16x16x32_bf16(false, a, false, b, (short)0,
                                                    acc, false, false);
    }
    int col = nt * 16 + (lane & 15);
    float bv = bias[col];
    int rbase = mt * 16 + ((lane < 16) ? 0 : 8);
#pragma unroll
    for (int r = 0; r < 8; ++r) {
      float v = acc[r] + bv;
      ys[(rbase + r) * COUT + col] = v > 0.f ? v : 0.f;
    }
  }
  __syncthreads();
  for (int c = tid; c < COUT; c += 256) {
    float s = 0.f, ss = 0.f;
#pragma unroll 8
    for (int r = 0; r < 32; ++r) {
      float v = ys[r * COUT + c];
      s += v;
      ss += v * v;
    }
    float m = s * (1.f / 32.f);
    float var = ss * (1.f / 32.f) - m * m;
    float sc = gam[c] * rsqrtf(var + EPSB);
    float sh = bet[c] - m * sc;
    for (int r = 0; r < 32; ++r) out[r * COUT + c] = ys[r * COUT + c] * sc + sh;
  }
}

// final 256->40 linear + row log_softmax
__global__ __launch_bounds__(64) void head3_kernel(const float* __restrict__ h2,
                                                   const float* __restrict__ W,
                                                   const float* __restrict__ bias,
                                                   float* __restrict__ out) {
  __shared__ float z[40];
  __shared__ float lse;
  int r = blockIdx.x, tid = threadIdx.x;
  if (tid < 40) {
    float acc = bias[tid];
    const float* hr = h2 + (size_t)r * 256;
    for (int c = 0; c < 256; ++c) acc += hr[c] * W[(size_t)c * 40 + tid];
    z[tid] = acc;
  }
  __syncthreads();
  if (tid == 0) {
    float m = -3.4e38f;
    for (int i = 0; i < 40; ++i) m = fmaxf(m, z[i]);
    float s = 0.f;
    for (int i = 0; i < 40; ++i) s += expf(z[i] - m);
    lse = m + logf(s);
  }
  __syncthreads();
  if (tid < 40) out[(size_t)r * 40 + tid] = z[tid] - lse;
}

// ---------------------------------------------------------------------------
extern "C" void kernel_launch(void* const* d_in, const int* in_sizes, int n_in,
                              void* d_out, int out_size, void* d_ws, size_t ws_size,
                              hipStream_t stream) {
  (void)in_sizes; (void)n_in; (void)out_size; (void)ws_size;
  // input order: pos, batch, conv1 L1..L3 {W,b,beta,g}, conv2 {W,b,beta,g},
  //              lin1 {W,b,beta,g}, head1, head2, head3 {W,b}
  const float* pos = (const float*)d_in[0];
  const float* c1W[3]    = {(const float*)d_in[2], (const float*)d_in[6], (const float*)d_in[10]};
  const float* c1b[3]    = {(const float*)d_in[3], (const float*)d_in[7], (const float*)d_in[11]};
  const float* c1beta[3] = {(const float*)d_in[4], (const float*)d_in[8], (const float*)d_in[12]};
  const float* c1g[3]    = {(const float*)d_in[5], (const float*)d_in[9], (const float*)d_in[13]};
  const float *c2W = (const float*)d_in[14], *c2b = (const float*)d_in[15],
              *c2beta = (const float*)d_in[16], *c2g = (const float*)d_in[17];
  const float *l1W = (const float*)d_in[18], *l1b = (const float*)d_in[19],
              *l1beta = (const float*)d_in[20], *l1g = (const float*)d_in[21];
  const float *h1W = (const float*)d_in[22], *h1b = (const float*)d_in[23],
              *h1beta = (const float*)d_in[24], *h1g = (const float*)d_in[25];
  const float *h2W = (const float*)d_in[26], *h2b = (const float*)d_in[27],
              *h2beta = (const float*)d_in[28], *h2g = (const float*)d_in[29];
  const float *h3W = (const float*)d_in[30], *h3b = (const float*)d_in[31];

  // ---- workspace layout ----
  char* base = (char*)d_ws;
  size_t off = 0;
  auto take = [&](size_t bytes) {
    void* p = base + off;
    off += (bytes + 1023) & ~(size_t)1023;
    return p;
  };
  // region A: dist matrix D (B*P*P f32) reused later as bf16 edge buffers e0/e1
  size_t szD = (size_t)BB * PP * PP * sizeof(float);
  size_t szE = MEDGE * 64 * sizeof(__bf16) * 2;
  float* Dm = (float*)take(szD > szE ? szD : szE);
  __bf16* e0 = (__bf16*)Dm;
  __bf16* e1 = e0 + MEDGE * 64;
  float* x1pre = (float*)take(BP * 64 * sizeof(float));
  float* x1    = (float*)take(BP * 64 * sizeof(float));
  float* x2pre = (float*)take(BP * 128 * sizeof(float));
  float* x2    = (float*)take(BP * 128 * sizeof(float));
  int* idx1 = (int*)take(MEDGE * sizeof(int));
  int* idx2 = (int*)take(MEDGE * sizeof(int));
  float* gpre = (float*)take(32 * 1024 * sizeof(float));
  float* gbn  = (float*)take(32 * 1024 * sizeof(float));
  float* h1buf = (float*)take(32 * 512 * sizeof(float));
  float* h2buf = (float*)take(32 * 256 * sizeof(float));
  __bf16* Wfrag = (__bf16*)take(192 * 1024 * sizeof(__bf16));
  float* bp  = (float*)take(1024 * sizeof(float));
  float* stats = (float*)take(2048 * sizeof(float));

  auto Z = [&](float* p, size_t n) {
    int g = (int)((n + 255) / 256);
    if (g > 4096) g = 4096;
    fill_zero<<<g, 256, 0, stream>>>(p, n);
  };
  const float fM = (float)MEDGE, fBP = (float)BP;

  // ---- knn on pos ----
  dist_kernel<3><<<dim3(PP / 16, BB), 256, 0, stream>>>(pos, Dm);
  knn_select<<<(int)(BP / 128), 128, 0, stream>>>(Dm, idx1);

  // ---- conv1 layer 1 (6->64) ----
  prep_bias<<<64, 64, 0, stream>>>(c1W[0], c1b[0], nullptr, nullptr, nullptr, 1.f,
                                   0, 6, 64, bp);
  prep_wfrag<<<8, 256, 0, stream>>>(c1W[0], nullptr, nullptr, 1.f, 0, 6, 32, 64,
                                    Wfrag);
  Z(stats, 2048);
  mlp_gemm<32, 64, 0, 0><<<(int)(MEDGE / 128), 256, 0, stream>>>(
      pos, nullptr, nullptr, idx1, Wfrag, bp, e0, stats, 1);

  // ---- conv1 layer 2 (64->64), BN1 folded ----
  prep_bias<<<64, 64, 0, stream>>>(c1W[1], c1b[1], c1g[0], c1beta[0], stats, fM,
                                   1, 64, 64, bp);
  prep_wfrag<<<16, 256, 0, stream>>>(c1W[1], c1g[0], stats, fM, 1, 64, 64, 64,
                                     Wfrag);
  Z(stats, 2048);
  mlp_gemm<64, 64, 1, 0><<<(int)(MEDGE / 128), 256, 0, stream>>>(
      nullptr, nullptr, e0, nullptr, Wfrag, bp, e1, stats, 1);

  // ---- conv1 layer 3 (64->64), BN2 folded, max over k fused ----
  prep_bias<<<64, 64, 0, stream>>>(c1W[2], c1b[2], c1g[1], c1beta[1], stats, fM,
                                   1, 64, 64, bp);
  prep_wfrag<<<16, 256, 0, stream>>>(c1W[2], c1g[1], stats, fM, 1, 64, 64, 64,
                                     Wfrag);
  Z(stats, 2048);
  Z(x1pre, BP * 64);
  mlp_gemm<64, 64, 1, 1><<<(int)(MEDGE / 128), 256, 0, stream>>>(
      nullptr, nullptr, e1, nullptr, Wfrag, bp, x1pre, stats, KNN);
  bn_apply<<<4096, 256, 0, stream>>>(x1pre, x1, stats, c1g[2], c1beta[2], fM, 64,
                                     BP * 64);

  // ---- knn on x1 ----
  dist_kernel<64><<<dim3(PP / 16, BB), 256, 0, stream>>>(x1, Dm);
  knn_select<<<(int)(BP / 128), 128, 0, stream>>>(Dm, idx2);

  // ---- conv2 (128->128), max over k fused ----
  prep_bias<<<128, 64, 0, stream>>>(c2W, c2b, nullptr, nullptr, nullptr, 1.f, 0,
                                    128, 128, bp);
  prep_wfrag<<<64, 256, 0, stream>>>(c2W, nullptr, nullptr, 1.f, 0, 128, 128, 128,
                                     Wfrag);
  Z(stats, 2048);
  Z(x2pre, BP * 128);
  mlp_gemm<128, 128, 2, 1><<<(int)(MEDGE / 128), 256, 0, stream>>>(
      x1, nullptr, nullptr, idx2, Wfrag, bp, x2pre, stats, KNN);
  bn_apply<<<4096, 256, 0, stream>>>(x2pre, x2, stats, c2g, c2beta, fM, 128,
                                     BP * 128);

  // ---- lin1 (192->1024), max over P fused ----
  prep_bias<<<1024, 64, 0, stream>>>(l1W, l1b, nullptr, nullptr, nullptr, 1.f, 0,
                                     192, 1024, bp);
  prep_wfrag<<<768, 256, 0, stream>>>(l1W, nullptr, nullptr, 1.f, 0, 192, 192,
                                      1024, Wfrag);
  Z(stats, 2048);
  Z(gpre, 32 * 1024);
  mlp_gemm<192, 1024, 3, 1><<<(int)(BP / 128), 256, 0, stream>>>(
      x1, x2, nullptr, nullptr, Wfrag, bp, gpre, stats, PP);
  bn_apply<<<128, 256, 0, stream>>>(gpre, gbn, stats, l1g, l1beta, fBP, 1024,
                                    32 * 1024);

  // ---- heads ----
  head_kernel<1024, 512><<<1, 256, 0, stream>>>(gbn, h1W, h1b, h1g, h1beta, h1buf);
  head_kernel<512, 256><<<1, 256, 0, stream>>>(h1buf, h2W, h2b, h2g, h2beta, h2buf);
  head3_kernel<<<32, 64, 0, stream>>>(h2buf, h3W, h3b, (float*)d_out);
}
